// GraphEncoder_58987080843443
// MI455X (gfx1250) — matrix operand order repaired and checked by reading, not compiled, not run on hardware
//
#include <hip/hip_runtime.h>
#include <math.h>

typedef __bf16 bf16;
typedef __attribute__((ext_vector_type(16))) bf16   v16bf;
typedef __attribute__((ext_vector_type(8)))  float  v8f;
typedef __attribute__((ext_vector_type(4)))  unsigned int u32x4;

#define N_ 256
#define D_ 512
#define L_ 5
#define E_ 65280   // N*(N-1)
#define R_ 255     // edges per subject
#define KK 512     // all GEMMs here have K = D_ = 512
#define STRIDE 520 // padded LDS row stride (bank-conflict free, 16B aligned)

union FragBF { v16bf v; u32x4 q[2]; };

// B-matrix 32x16 bf16 fragment where B = W^T and W is row-major NxK.
// lanes 0-15 -> N=lane, K={0..7,16..23} ; lanes 16-31 -> K shifted +8 (matches A layout)
__device__ inline v16bf load_b_frag(const bf16* W, int ldw, int nBase, int k0, int lane) {
    int hi = (lane >> 4) & 1;
    const bf16* p = W + (long)(nBase + (lane & 15)) * ldw + k0 + hi * 8;
    FragBF f;
    f.q[0] = *(const u32x4*)(p);
    f.q[1] = *(const u32x4*)(p + 16);
    return f.v;
}

// A-matrix 16x32 bf16 fragment from an LDS panel [16][STRIDE].
// ISA layout: lanes 0-15 -> M=lane, K={0..7,16..23}; lanes 16-31 -> K shifted +8.
__device__ inline v16bf lds_a_frag(const bf16* panel, int k0, int lane) {
    int hi = (lane >> 4) & 1;
    const bf16* p = panel + (lane & 15) * STRIDE + k0 + hi * 8;
    FragBF f;
    f.q[0] = *(const u32x4*)(p);        // K = base+0..7
    f.q[1] = *(const u32x4*)(p + 16);   // K = base+16..23
    return f.v;
}

// Cooperative copy of a 16 x 512 bf16 row-panel into LDS (padded stride).
// 256 threads x 32 elements = 8192 elements = exactly one pass.
__device__ inline void stage_panel(bf16* panel, const bf16* A, int mBase) {
    int f = threadIdx.x * 32;
    int r = f >> 9, k = f & (KK - 1);
    const bf16* src = A + (long)(mBase + r) * KK + k;
    bf16* dst = panel + r * STRIDE + k;
    #pragma unroll
    for (int q = 0; q < 4; ++q)
        *(u32x4*)(dst + q * 8) = *(const u32x4*)(src + q * 8);
}

__device__ inline float sigmoidf_(float x) { return 1.0f / (1.0f + expf(-x)); }

// ---------------- generic GEMM: C = act(A @ W^T + bias*bscale) --------------
// A: Mx512 bf16 row-major; W: Nx512 bf16 row-major; outputs f32 and/or bf16.
// Block: 8 waves share one 16-row A panel in LDS; wave w -> N-tile blockIdx.y*8+w.
__global__ void gemm_bf16_kernel(const bf16* __restrict__ A, const bf16* __restrict__ W,
                                 const float* __restrict__ bias, float bscale,
                                 float* __restrict__ Cf, bf16* __restrict__ Cbf,
                                 int N, int relu) {
    alignas(16) __shared__ bf16 As[16 * STRIDE];
    int lane = threadIdx.x & 31;
    int wave = threadIdx.x >> 5;
    int tm = blockIdx.x;                     // row tile (16 rows)
    int tn = blockIdx.y * 8 + wave;          // column tile (16 cols), wave-uniform

    stage_panel(As, A, tm * 16);
    __syncthreads();

    v8f acc = {};
    for (int k0 = 0; k0 < KK; k0 += 32) {
        v16bf a = lds_a_frag(As, k0, lane);
        v16bf b = load_b_frag(W, KK, tn * 16, k0, lane);
        acc = __builtin_amdgcn_wmma_f32_16x16x32_bf16(false, a, false, b,
                                                      (short)0, acc, false, false);
    }
    int hi  = (lane >> 4) & 1;
    int col = tn * 16 + (lane & 15);
    float bv = bias ? bias[col] * bscale : 0.0f;
    #pragma unroll
    for (int v = 0; v < 8; ++v) {
        int row = tm * 16 + v + 8 * hi;
        float x = acc[v] + bv;
        if (relu) x = fmaxf(x, 0.0f);
        long o = (long)row * N + col;
        if (Cf)  Cf[o]  = x;
        if (Cbf) Cbf[o] = (bf16)x;
    }
}

// ------------- fused edge GRU: he2 = GRU(Mp, he) + she accumulate -----------
// 6 f32 accumulators per wave (r/z/n gates x {ih,hh}); Mp/he panels in LDS,
// shared by 8 waves (each covering a different 16-col tile of D).
__global__ void edge_gru_kernel(const bf16* __restrict__ Mp, const bf16* __restrict__ Hb,
                                const float* __restrict__ Hold,
                                const bf16* __restrict__ Wih, const bf16* __restrict__ Whh,
                                const float* __restrict__ bih, const float* __restrict__ bhh,
                                float* __restrict__ Hnew, bf16* __restrict__ Hnb,
                                float* __restrict__ She, int first) {
    alignas(16) __shared__ bf16 Ms[16 * STRIDE];
    alignas(16) __shared__ bf16 Hs[16 * STRIDE];
    int lane = threadIdx.x & 31;
    int wave = threadIdx.x >> 5;
    int tm = blockIdx.x;                     // edge-row tile (16 rows of E)
    int tn = blockIdx.y * 8 + wave;          // col tile of D

    stage_panel(Ms, Mp, tm * 16);
    stage_panel(Hs, Hb, tm * 16);
    __syncthreads();

    v8f a_ir = {}, a_iz = {}, a_in = {}, a_hr = {}, a_hz = {}, a_hn = {};
    for (int k0 = 0; k0 < KK; k0 += 32) {
        v16bf am = lds_a_frag(Ms, k0, lane);
        v16bf ah = lds_a_frag(Hs, k0, lane);
        v16bf br = load_b_frag(Wih, KK, tn * 16 + 0 * D_, k0, lane);
        v16bf bz = load_b_frag(Wih, KK, tn * 16 + 1 * D_, k0, lane);
        v16bf bn = load_b_frag(Wih, KK, tn * 16 + 2 * D_, k0, lane);
        v16bf cr = load_b_frag(Whh, KK, tn * 16 + 0 * D_, k0, lane);
        v16bf cz = load_b_frag(Whh, KK, tn * 16 + 1 * D_, k0, lane);
        v16bf cn = load_b_frag(Whh, KK, tn * 16 + 2 * D_, k0, lane);
        a_ir = __builtin_amdgcn_wmma_f32_16x16x32_bf16(false, am, false, br, (short)0, a_ir, false, false);
        a_iz = __builtin_amdgcn_wmma_f32_16x16x32_bf16(false, am, false, bz, (short)0, a_iz, false, false);
        a_in = __builtin_amdgcn_wmma_f32_16x16x32_bf16(false, am, false, bn, (short)0, a_in, false, false);
        a_hr = __builtin_amdgcn_wmma_f32_16x16x32_bf16(false, ah, false, cr, (short)0, a_hr, false, false);
        a_hz = __builtin_amdgcn_wmma_f32_16x16x32_bf16(false, ah, false, cz, (short)0, a_hz, false, false);
        a_hn = __builtin_amdgcn_wmma_f32_16x16x32_bf16(false, ah, false, cn, (short)0, a_hn, false, false);
    }
    int hi  = (lane >> 4) & 1;
    int col = tn * 16 + (lane & 15);
    float bir = bih[col], biz = bih[col + D_], bin = bih[col + 2 * D_];
    float bhr = bhh[col], bhz = bhh[col + D_], bhn = bhh[col + 2 * D_];
    #pragma unroll
    for (int v = 0; v < 8; ++v) {
        int row = tm * 16 + v + 8 * hi;
        long o = (long)row * D_ + col;
        float rg = sigmoidf_((a_ir[v] + bir) + (a_hr[v] + bhr));
        float z  = sigmoidf_((a_iz[v] + biz) + (a_hz[v] + bhz));
        float nn = tanhf((a_in[v] + bin) + rg * (a_hn[v] + bhn));
        float hp = Hold[o];
        float h2 = (1.0f - z) * nn + z * hp;
        Hnew[o] = h2;
        Hnb[o]  = (bf16)h2;
        She[o]  = first ? h2 : (She[o] + h2);
    }
}

// --------- node GRU elementwise (gi, gh precomputed with biases) ------------
__global__ void gru_node_kernel(const float* __restrict__ gi, const float* __restrict__ gh,
                                const float* __restrict__ Hold,
                                float* __restrict__ Hnew, bf16* __restrict__ Hnb,
                                float* __restrict__ Sh, int first) {
    int i = blockIdx.x * blockDim.x + threadIdx.x;
    if (i >= N_ * D_) return;
    int row = i >> 9, col = i & (D_ - 1);
    long b = (long)row * (3 * D_);
    float ir = gi[b + col], iz = gi[b + D_ + col], inn = gi[b + 2 * D_ + col];
    float hr = gh[b + col], hz = gh[b + D_ + col], hn  = gh[b + 2 * D_ + col];
    float rg = sigmoidf_(ir + hr);
    float z  = sigmoidf_(iz + hz);
    float nn = tanhf(inn + rg * hn);
    float h2 = (1.0f - z) * nn + z * Hold[i];
    Hnew[i] = h2;
    Hnb[i]  = (bf16)h2;
    Sh[i]   = first ? h2 : (Sh[i] + h2);
}

// ---- edge message: Mp = LN(Ps[s] + Po[o]) -> bf16 ; one wave per edge row --
__global__ void edge_msg_kernel(const float* __restrict__ Ps, const float* __restrict__ Po,
                                const int* __restrict__ ei,
                                const float* __restrict__ g, const float* __restrict__ b,
                                bf16* __restrict__ Mp) {
    int lane = threadIdx.x & 31;
    int row  = blockIdx.x * (blockDim.x >> 5) + (threadIdx.x >> 5);
    if (row >= E_) return;
    int s = ei[row], o = ei[E_ + row];
    const float* ps = Ps + (long)s * D_;
    const float* po = Po + (long)o * D_;
    float x[16], sum = 0.0f, sq = 0.0f;
    #pragma unroll
    for (int j = 0; j < 16; ++j) {
        int d = lane + 32 * j;
        float v = ps[d] + po[d];
        x[j] = v; sum += v; sq += v * v;
    }
    for (int off = 16; off > 0; off >>= 1) {
        sum += __shfl_xor(sum, off, 32);
        sq  += __shfl_xor(sq,  off, 32);
    }
    float m  = sum * (1.0f / D_);
    float va = sq * (1.0f / D_) - m * m;
    float rs = rsqrtf(va + 1e-5f);
    long base = (long)row * D_;
    #pragma unroll
    for (int j = 0; j < 16; ++j) {
        int d = lane + 32 * j;
        Mp[base + d] = (bf16)((x[j] - m) * rs * g[d] + b[d]);
    }
}

// -------- in-place row LayerNorm (final outputs) ----------------------------
__global__ void row_ln_kernel(float* __restrict__ X, const float* __restrict__ g,
                              const float* __restrict__ b, int rows) {
    int lane = threadIdx.x & 31;
    int row  = blockIdx.x * (blockDim.x >> 5) + (threadIdx.x >> 5);
    if (row >= rows) return;
    float* p = X + (long)row * D_;
    float x[16], sum = 0.0f, sq = 0.0f;
    #pragma unroll
    for (int j = 0; j < 16; ++j) {
        int d = lane + 32 * j;
        float v = p[d];
        x[j] = v; sum += v; sq += v * v;
    }
    for (int off = 16; off > 0; off >>= 1) {
        sum += __shfl_xor(sum, off, 32);
        sq  += __shfl_xor(sq,  off, 32);
    }
    float m  = sum * (1.0f / D_);
    float va = sq * (1.0f / D_) - m * m;
    float rs = rsqrtf(va + 1e-5f);
    #pragma unroll
    for (int j = 0; j < 16; ++j) {
        int d = lane + 32 * j;
        p[d] = (x[j] - m) * rs * g[d] + b[d];
    }
}

// -------- per-subject sum of ReLU activations: S1[n] = sum_j X1[n*255+j] ----
__global__ void subj_reduce_kernel(const bf16* __restrict__ X1, bf16* __restrict__ S1) {
    int i = blockIdx.x * blockDim.x + threadIdx.x;
    if (i >= N_ * D_) return;
    int n = i >> 9, d = i & (D_ - 1);
    const bf16* p = X1 + (long)n * R_ * D_ + d;
    float acc = 0.0f;
    for (int j = 0; j < R_; ++j) acc += (float)p[(long)j * D_];
    S1[i] = (bf16)acc;
}

// -------- f32 -> bf16 pack --------------------------------------------------
__global__ void pack_bf16_kernel(const float* __restrict__ s, bf16* __restrict__ d, long n) {
    long i = (long)blockIdx.x * blockDim.x + threadIdx.x;
    if (i < n) d[i] = (bf16)s[i];
}

// ============================================================================
extern "C" void kernel_launch(void* const* d_in, const int* in_sizes, int n_in,
                              void* d_out, int out_size, void* d_ws, size_t ws_size,
                              hipStream_t stream) {
    (void)in_sizes; (void)n_in; (void)out_size; (void)ws_size;
    const float* h0       = (const float*)d_in[0];
    const float* he0      = (const float*)d_in[1];
    const float* phis_W1  = (const float*)d_in[2];
    const float* phis_b1  = (const float*)d_in[3];
    const float* phis_W2  = (const float*)d_in[4];
    const float* phis_b2  = (const float*)d_in[5];
    const float* phio_W1  = (const float*)d_in[6];
    const float* phio_b1  = (const float*)d_in[7];
    const float* phio_W2  = (const float*)d_in[8];
    const float* phio_b2  = (const float*)d_in[9];
    const float* phip_W1  = (const float*)d_in[10];
    const float* phip_b1  = (const float*)d_in[11];
    const float* phip_W2  = (const float*)d_in[12];
    const float* phip_b2  = (const float*)d_in[13];
    const float* ln_g     = (const float*)d_in[14];
    const float* ln_b     = (const float*)d_in[15];
    const float* grun_Wih = (const float*)d_in[16];
    const float* grun_Whh = (const float*)d_in[17];
    const float* grun_bih = (const float*)d_in[18];
    const float* grun_bhh = (const float*)d_in[19];
    const float* grue_Wih = (const float*)d_in[20];
    const float* grue_Whh = (const float*)d_in[21];
    const float* grue_bih = (const float*)d_in[22];
    const float* grue_bhh = (const float*)d_in[23];
    const float* fln_g    = (const float*)d_in[24];
    const float* fln_b    = (const float*)d_in[25];
    const int*   ei       = (const int*)  d_in[26];

    float* out = (float*)d_out;
    float* sh  = out;                       // N*D running sum (accumulated in-place)
    float* she = out + (size_t)N_ * D_;     // E*D running sum

    // ---- bump allocator over workspace ----
    char* wp = (char*)d_ws;
    auto alloc = [&](size_t bytes) -> char* {
        char* r = wp; wp += (bytes + 255) & ~(size_t)255; return r;
    };
    const size_t DD = (size_t)D_ * D_, DD3 = 3 * DD;
    bf16* wsW1  = (bf16*)alloc(L_ * DD  * 2);
    bf16* wsW2  = (bf16*)alloc(L_ * DD  * 2);
    bf16* woW1  = (bf16*)alloc(L_ * DD  * 2);
    bf16* woW2  = (bf16*)alloc(L_ * DD  * 2);
    bf16* wpW1  = (bf16*)alloc(L_ * DD  * 2);
    bf16* wpW2  = (bf16*)alloc(L_ * DD  * 2);
    bf16* wnWih = (bf16*)alloc(L_ * DD3 * 2);
    bf16* wnWhh = (bf16*)alloc(L_ * DD3 * 2);
    bf16* weWih = (bf16*)alloc(L_ * DD3 * 2);
    bf16* weWhh = (bf16*)alloc(L_ * DD3 * 2);

    float* hBuf[2]  = { (float*)alloc((size_t)N_ * D_ * 4), (float*)alloc((size_t)N_ * D_ * 4) };
    bf16*  hbBuf[2] = { (bf16*) alloc((size_t)N_ * D_ * 2), (bf16*) alloc((size_t)N_ * D_ * 2) };
    float* heBuf[2] = { (float*)alloc((size_t)E_ * D_ * 4), (float*)alloc((size_t)E_ * D_ * 4) };
    bf16*  hebBuf[2]= { (bf16*) alloc((size_t)E_ * D_ * 2), (bf16*) alloc((size_t)E_ * D_ * 2) };

    bf16*  MpB  = (bf16*) alloc((size_t)E_ * D_ * 2);
    bf16*  X1B  = (bf16*) alloc((size_t)E_ * D_ * 2);
    bf16*  S1B  = (bf16*) alloc((size_t)N_ * D_ * 2);
    bf16*  tmpB = (bf16*) alloc((size_t)N_ * D_ * 2);
    float* Ps   = (float*)alloc((size_t)N_ * D_ * 4);
    float* Po   = (float*)alloc((size_t)N_ * D_ * 4);
    float* Mn   = (float*)alloc((size_t)N_ * D_ * 4);
    bf16*  MnB  = (bf16*) alloc((size_t)N_ * D_ * 2);
    float* giN  = (float*)alloc((size_t)N_ * 3 * D_ * 4);
    float* ghN  = (float*)alloc((size_t)N_ * 3 * D_ * 4);

    const int TB = 256;
    auto packN = [&](const float* s, bf16* d, long n) {
        pack_bf16_kernel<<<dim3((unsigned)((n + TB - 1) / TB)), dim3(TB), 0, stream>>>(s, d, n);
    };
    auto gemm = [&](const bf16* A, const bf16* W, const float* bias, float bscale,
                    float* Cf, bf16* Cbf, int M, int N, int relu) {
        gemm_bf16_kernel<<<dim3((unsigned)(M / 16), (unsigned)(N / 128)), dim3(TB), 0, stream>>>(
            A, W, bias, bscale, Cf, Cbf, N, relu);
    };

    // ---- pack all weights to bf16 (once) ----
    packN(phis_W1, wsW1,  (long)L_ * DD);
    packN(phis_W2, wsW2,  (long)L_ * DD);
    packN(phio_W1, woW1,  (long)L_ * DD);
    packN(phio_W2, woW2,  (long)L_ * DD);
    packN(phip_W1, wpW1,  (long)L_ * DD);
    packN(phip_W2, wpW2,  (long)L_ * DD);
    packN(grun_Wih, wnWih, (long)L_ * DD3);
    packN(grun_Whh, wnWhh, (long)L_ * DD3);
    packN(grue_Wih, weWih, (long)L_ * DD3);
    packN(grue_Whh, weWhh, (long)L_ * DD3);

    // ---- pack initial states ----
    packN(h0,  hbBuf[0],  (long)N_ * D_);
    packN(he0, hebBuf[0], (long)E_ * D_);

    const float* hc  = h0;       const bf16* hbc  = hbBuf[0];
    const float* hec = he0;      const bf16* hebc = hebBuf[0];
    int cur = 0;

    for (int l = 0; l < L_; ++l) {
        int wr = cur ^ 1;
        const bf16*  sW1 = wsW1 + (size_t)l * DD;  const float* sb1 = phis_b1 + (size_t)l * D_;
        const bf16*  sW2 = wsW2 + (size_t)l * DD;  const float* sb2 = phis_b2 + (size_t)l * D_;
        const bf16*  oW1 = woW1 + (size_t)l * DD;  const float* ob1 = phio_b1 + (size_t)l * D_;
        const bf16*  oW2 = woW2 + (size_t)l * DD;  const float* ob2 = phio_b2 + (size_t)l * D_;
        const bf16*  pW1 = wpW1 + (size_t)l * DD;  const float* pb1 = phip_b1 + (size_t)l * D_;
        const bf16*  pW2 = wpW2 + (size_t)l * DD;  const float* pb2 = phip_b2 + (size_t)l * D_;
        const float* lg  = ln_g + (size_t)l * D_;  const float* lb  = ln_b + (size_t)l * D_;
        const bf16*  nWih = wnWih + (size_t)l * DD3; const float* nbih = grun_bih + (size_t)l * 3 * D_;
        const bf16*  nWhh = wnWhh + (size_t)l * DD3; const float* nbhh = grun_bhh + (size_t)l * 3 * D_;
        const bf16*  eWih = weWih + (size_t)l * DD3; const float* ebih = grue_bih + (size_t)l * 3 * D_;
        const bf16*  eWhh = weWhh + (size_t)l * DD3; const float* ebhh = grue_bhh + (size_t)l * 3 * D_;

        // Ps = phis(h) on distinct nodes (2 small GEMMs), Po = phio(h)
        gemm(hbc,  sW1, sb1, 1.0f, nullptr, tmpB, N_, D_, 1);
        gemm(tmpB, sW2, sb2, 1.0f, Ps, nullptr,   N_, D_, 0);
        gemm(hbc,  oW1, ob1, 1.0f, nullptr, tmpB, N_, D_, 1);
        gemm(tmpB, oW2, ob2, 1.0f, Po, nullptr,   N_, D_, 0);

        // Mp = LN(Ps[s] + Po[o]) -> bf16, E rows
        edge_msg_kernel<<<E_ / 8, TB, 0, stream>>>(Ps, Po, ei, lg, lb, MpB);

        // X1 = relu(he @ pW1^T + pb1)  (big E-scale GEMM, LDS-staged A)
        gemm(hebc, pW1, pb1, 1.0f, nullptr, X1B, E_, D_, 1);
        // S1[n] = sum over 255 edges; second linear collapsed to N-scale:
        subj_reduce_kernel<<<(N_ * D_) / TB, TB, 0, stream>>>(X1B, S1B);
        gemm(S1B, pW2, pb2, (float)R_, Mn, MnB, N_, D_, 0);

        // node GRU: gi = Mn @ nWih^T + bih ; gh = h @ nWhh^T + bhh ; elementwise
        gemm(MnB, nWih, nbih, 1.0f, giN, nullptr, N_, 3 * D_, 0);
        gemm(hbc, nWhh, nbhh, 1.0f, ghN, nullptr, N_, 3 * D_, 0);
        gru_node_kernel<<<(N_ * D_) / TB, TB, 0, stream>>>(
            giN, ghN, hc, hBuf[wr], hbBuf[wr], sh, l == 0);

        // fused edge GRU (dominant compute: 6 WMMA accumulators / tile,
        // Mp/he panels shared via LDS across 8 waves)
        edge_gru_kernel<<<dim3(E_ / 16, D_ / 128), TB, 0, stream>>>(
            MpB, hebc, hec, eWih, eWhh, ebih, ebhh,
            heBuf[wr], hebBuf[wr], she, l == 0);

        hc = hBuf[wr]; hbc = hbBuf[wr];
        hec = heBuf[wr]; hebc = hebBuf[wr];
        cur = wr;
    }

    // final LayerNorms, in place in d_out
    row_ln_kernel<<<N_ / 8, TB, 0, stream>>>(sh,  fln_g, fln_b, N_);
    row_ln_kernel<<<E_ / 8, TB, 0, stream>>>(she, fln_g, fln_b, E_);
}
